// SnnResidualBlockM_40157944217696
// MI455X (gfx1250) — matrix operand order repaired and checked by reading, not compiled, not run on hardware
//
#include <hip/hip_runtime.h>
#include <hip/hip_bf16.h>
#include <cstdint>

// ---------------------------------------------------------------------------
// SNN residual block for MI455X (gfx1250, wave32, WMMA).
// Shapes: B=4, T=8, C=128, C2=64, H=W=80, HW=6400, TB=T*B=32, NPIX=TB*HW.
// All convs run as f16 WMMA GEMMs with f32 accumulation.
// The lcb1 mem-conv (m1w) is dead in the reference and is skipped entirely.
// ---------------------------------------------------------------------------

typedef __attribute__((ext_vector_type(16))) _Float16 v16h;
typedef __attribute__((ext_vector_type(8)))  _Float16 v8h;
typedef __attribute__((ext_vector_type(8)))  float    v8f;

#define B_    4
#define T_    8
#define C1_   128
#define C2_   64
#define H_    80
#define W_    80
#define HW_   6400
#define TB_   32
#define NPIX_ 204800    // TB_*HW_
#define EPS_  1e-5f

// Workspace layout (bytes)
#define OFF_SPK1 ((size_t)0)           // NPIX*128 f16   = 52,428,800
#define OFF_S1   ((size_t)52428800)    // NPIX*64  f32   = 52,428,800
#define OFF_SPK2 ((size_t)104857600)   // NPIX*64  f16   = 26,214,400
#define OFF_MEM2 ((size_t)131072000)   // NPIX*64  f16   = 26,214,400
#define OFF_WC1  ((size_t)157286400)   // 64*128   f16
#define OFF_WC2  ((size_t)157302784)   // 128*9*64 f16
#define OFF_WM2  ((size_t)157450240)   // 128*9*64 f16
#define OFF_ST1  ((size_t)157597696)   // 512*2 f32
#define OFF_ST2  ((size_t)157601792)   // 1024*2 f32

__device__ inline v8f wmma_f16(v16h a, v16h b, v8f c) {
    // v_wmma_f32_16x16x32_f16  (8-arg form, probe-confirmed)
    return __builtin_amdgcn_wmma_f32_16x16x32_f16(
        /*neg_a=*/false, a, /*neg_b=*/false, b,
        /*c_mod=*/(short)0, c, /*reuse_a=*/false, /*reuse_b=*/false);
}

// A-fragment: lane holds row m = lane%16; halves 0..15 = K at (kg*16 .. +15)
// within a 32-wide K step (kg = lane/16).
__device__ inline v16h load_afrag(const _Float16* w) {
    v8h lo = *(const v8h*)(w);
    v8h hi = *(const v8h*)(w + 8);
    v16h a;
#pragma unroll
    for (int i = 0; i < 8; ++i) { a[i] = lo[i]; a[i + 8] = hi[i]; }
    return a;
}

// ---------------------------------------------------------------------------
// K0: weight prep. Wc1: [64][128] f16. Wc2/Wm2: [co][tap][ci] f16 (tap-major)
// so implicit-GEMM K runs contiguously per 3x3 tap.
// ---------------------------------------------------------------------------
__global__ void k_prep_weights(const float* __restrict__ c1w,
                               const float* __restrict__ c2w,
                               const float* __restrict__ m2w,
                               _Float16* __restrict__ Wc1,
                               _Float16* __restrict__ Wc2,
                               _Float16* __restrict__ Wm2) {
    int i = blockIdx.x * 256 + threadIdx.x;
    if (i < 64 * 128) Wc1[i] = (_Float16)c1w[i];
    if (i < 128 * 9 * 64) {
        int co = i / 576, r = i % 576;
        int tap = r / 64, ci = r % 64;
        int src = co * 576 + ci * 9 + tap;   // [co][ci][ky][kx]
        Wc2[i] = (_Float16)c2w[src];
        Wm2[i] = (_Float16)m2w[src];
    }
}

// ---------------------------------------------------------------------------
// K1: LIF1.  x:[B][T][128][6400] f32 -> spk1 pixel-major [(t*B+b)*6400+p][128] f16.
// Per-thread T-sequential recurrence; LDS tile transpose for coalesced stores.
// ---------------------------------------------------------------------------
__global__ void k_lif1(const float* __restrict__ x,
                       const float* __restrict__ pbeta,
                       const float* __restrict__ pth,
                       _Float16* __restrict__ spk1) {
    __shared__ _Float16 tile[16][17];
    int tid = threadIdx.x;
    int bi = blockIdx.x;                    // 12800 = B * 8 ctiles * 400 ptiles
    int ptile = bi % 400;
    int ctile = (bi / 400) & 7;
    int b     = bi / 3200;
    int p0 = ptile * 16, c0 = ctile * 16;
    int pp = tid & 15, cc = tid >> 4;       // compute-phase: p fastest (coalesced x)
    int c2 = tid & 15, p2 = tid >> 4;       // store-phase:   c fastest (coalesced spk1)

    float beta = fminf(fmaxf(pbeta[0], 0.f), 1.f);
    float th   = pth[0];
    float mem  = 0.f;
#pragma unroll
    for (int t = 0; t < T_; ++t) {
        float xv = x[((size_t)(b * T_ + t) * C1_ + (c0 + cc)) * HW_ + p0 + pp];
        float reset = (mem - th > 0.f) ? 1.f : 0.f;   // detached spike of prev mem
        mem = beta * mem + xv - reset * th;
        tile[pp][cc] = (mem - th > 0.f) ? (_Float16)1.f : (_Float16)0.f;
        __syncthreads();
        spk1[((size_t)((t * B_ + b) * HW_ + p0 + p2)) * C1_ + c0 + c2] = tile[p2][c2];
        __syncthreads();
    }
}

// ---------------------------------------------------------------------------
// K2: 1x1 conv c1 as WMMA GEMM. M=64 (Cout), K=128 (Cin), N=NPIX.
// Block: 8 waves = 4 M-tiles x 2 N-subtiles over a 32-pixel tile staged in LDS.
// ---------------------------------------------------------------------------
__global__ void k_conv1x1_wmma(const _Float16* __restrict__ spk1,
                               const _Float16* __restrict__ Wc1,
                               const float* __restrict__ bias,
                               float* __restrict__ s1) {
    __shared__ __align__(32) _Float16 sAct[32 * 128];   // 8 KB
    int tid = threadIdx.x;
    int g0 = blockIdx.x * 32;                           // 6400 blocks

    const uint32_t* src = (const uint32_t*)(spk1 + (size_t)g0 * 128);
    uint32_t* dst = (uint32_t*)sAct;
#pragma unroll
    for (int it = 0; it < 8; ++it)                      // 2048 dwords cooperative
        dst[tid + it * 256] = src[tid + it * 256];
    __syncthreads();

    int lane = tid & 31, wv = tid >> 5;
    int m0   = (wv & 3) * 16;                           // Cout tile
    int nsub = wv >> 2;                                 // 0..1
    int n = lane & 15, kg = lane >> 4;
    int pl = nsub * 16 + n;                             // local pixel 0..31

    v8f acc = {};
#pragma unroll
    for (int ks = 0; ks < 4; ++ks) {                    // K = 128 = 4 x 32
        v16h bfrag = *(const v16h*)&sAct[pl * 128 + ks * 32 + kg * 16];
        v16h afrag = load_afrag(&Wc1[(m0 + n) * 128 + ks * 32 + kg * 16]);
        acc = wmma_f16(afrag, bfrag, acc);
    }
    v8f vb = *(const v8f*)&bias[m0 + kg * 8];
    acc = acc + vb;
    *(v8f*)&s1[((size_t)(g0 + pl)) * 64 + m0 + kg * 8] = acc;
}

// ---------------------------------------------------------------------------
// K3: BN1 stats per (t, c) over (B,HW) = 25600 elems.  512 groups.
// ---------------------------------------------------------------------------
__global__ void k_bnstats1(const float* __restrict__ s1, float* __restrict__ stats) {
    int grp = blockIdx.x;                // t*64 + c
    int t = grp >> 6, c = grp & 63;
    int tid = threadIdx.x;
    float sum = 0.f, sq = 0.f;
    for (int i = tid; i < 25600; i += 256) {
        int b = i / HW_, p = i % HW_;
        float v = s1[((size_t)((t * B_ + b) * HW_ + p)) * 64 + c];
        sum += v; sq += v * v;
    }
    __shared__ float ss[256], s2[256];
    ss[tid] = sum; s2[tid] = sq; __syncthreads();
    for (int off = 128; off > 0; off >>= 1) {
        if (tid < off) { ss[tid] += ss[tid + off]; s2[tid] += s2[tid + off]; }
        __syncthreads();
    }
    if (tid == 0) {
        float mean = ss[0] / 25600.f;
        float var  = s2[0] / 25600.f - mean * mean;
        stats[grp * 2 + 0] = mean;
        stats[grp * 2 + 1] = rsqrtf(var + EPS_);
    }
}

// ---------------------------------------------------------------------------
// K4: LIF2 with fused BN1 normalize.  s1 -> spk2/mem2 f16 pixel-major.
// ---------------------------------------------------------------------------
__global__ void k_lif2(const float* __restrict__ s1,
                       const float* __restrict__ stats,
                       const float* __restrict__ g1,
                       const float* __restrict__ pbeta,
                       const float* __restrict__ pth,
                       _Float16* __restrict__ spk2,
                       _Float16* __restrict__ mem2) {
    int tid = threadIdx.x;
    int cc = tid & 63, pp = tid >> 6;      // c fastest: fully coalesced
    int bi = blockIdx.x;                   // 6400 = B * 1600
    int p  = (bi % 1600) * 4 + pp;
    int b  = bi / 1600;
    float beta = fminf(fmaxf(pbeta[0], 0.f), 1.f);
    float th   = pth[0];
    float mem  = 0.f;
#pragma unroll
    for (int t = 0; t < T_; ++t) {
        size_t g = (size_t)(t * B_ + b) * HW_ + p;
        float v    = s1[g * 64 + cc];
        float mean = stats[(t * 64 + cc) * 2 + 0];
        float inv  = stats[(t * 64 + cc) * 2 + 1];
        float h    = (v - mean) * inv * g1[t * 64 + cc];
        float reset = (mem - th > 0.f) ? 1.f : 0.f;
        mem = beta * mem + h - reset * th;
        spk2[g * 64 + cc] = (mem - th > 0.f) ? (_Float16)1.f : (_Float16)0.f;
        mem2[g * 64 + cc] = (_Float16)mem;
    }
}

// ---------------------------------------------------------------------------
// K5: 3x3 conv as implicit WMMA GEMM. M=128, K=576 (9 taps x 64ch), N=NPIX.
// Block: 16-pixel row tile + zero halo staged in LDS [3][18][64] f16;
// 8 waves cover all 8 Cout tiles.  Output written planar [*, c, p]:
//   permuteBT=0 : [(t*B+b)*128 + c][p]    (men, [T,B,C,H,W])
//   permuteBT=1 : [(b*T+t)*128 + c][p]    (s2 staged into out, [B,T,C,H,W])
// ---------------------------------------------------------------------------
__global__ void k_conv3x3_wmma(const _Float16* __restrict__ act,
                               const _Float16* __restrict__ Wt,
                               const float* __restrict__ bias,
                               float* __restrict__ out,
                               int permuteBT) {
    __shared__ __align__(32) _Float16 sB[3 * 18 * 64];  // 6912 B
    int tid = threadIdx.x;
    int g0 = blockIdx.x * 16;                           // 12800 blocks
    int tb = g0 / HW_;
    int p0 = g0 % HW_;
    int h0 = p0 / W_, w0 = p0 % W_;

    // Cooperative halo fill: 1728 dwords, OOB -> 0 (zero padding).
    uint32_t* dst = (uint32_t*)sB;
    const uint32_t* srcbase = (const uint32_t*)act;
#pragma unroll
    for (int it = 0; it < 7; ++it) {
        int slot = tid + it * 256;
        if (slot < 1728) {
            int cip = slot & 31;                        // dword pair of channels
            int wc  = (slot >> 5) % 18;
            int hr  = slot / (32 * 18);
            int hh = h0 + hr - 1, ww = w0 + wc - 1;
            uint32_t val = 0u;
            if (hh >= 0 && hh < H_ && ww >= 0 && ww < W_)
                val = srcbase[((size_t)tb * HW_ + hh * W_ + ww) * 32 + cip];
            dst[slot] = val;
        }
    }
    __syncthreads();

    int lane = tid & 31, wv = tid >> 5;
    int m0 = wv * 16;                                   // Cout tile
    int n = lane & 15, kg = lane >> 4;

    v8f acc = {};
#pragma unroll
    for (int tap = 0; tap < 9; ++tap) {
        int ty = tap / 3, tx = tap % 3;
        const _Float16* bbase = &sB[(ty * 18 + (n + tx)) * 64];
        const _Float16* wbase = &Wt[((size_t)(m0 + n) * 9 + tap) * 64];
#pragma unroll
        for (int hk = 0; hk < 2; ++hk) {                // 64 ch = 2 x 32 K-steps
            v16h bfrag = *(const v16h*)(bbase + hk * 32 + kg * 16);
            v16h afrag = load_afrag(wbase + hk * 32 + kg * 16);
            acc = wmma_f16(afrag, bfrag, acc);
        }
    }

    int tbo = permuteBT ? ((tb & 3) * T_ + (tb >> 2)) : tb;   // tb = t*B+b
#pragma unroll
    for (int j = 0; j < 8; ++j) {
        int c = m0 + kg * 8 + j;                        // D layout: M = 8*(l/16)+j
        out[((size_t)tbo * 128 + c) * HW_ + p0 + n] = acc[j] + bias[c];
    }
}

// ---------------------------------------------------------------------------
// K6: BN2 stats per (t, c) over (B,HW); s2 staged in out region [B,T,C,HW].
// ---------------------------------------------------------------------------
__global__ void k_bnstats2(const float* __restrict__ s2, float* __restrict__ stats) {
    int grp = blockIdx.x;                // t*128 + c  (1024 groups)
    int t = grp >> 7, c = grp & 127;
    int tid = threadIdx.x;
    float sum = 0.f, sq = 0.f;
    for (int i = tid; i < 25600; i += 256) {
        int b = i / HW_, p = i % HW_;
        float v = s2[((size_t)((b * T_ + t) * 128 + c)) * HW_ + p];
        sum += v; sq += v * v;
    }
    __shared__ float ss[256], s2l[256];
    ss[tid] = sum; s2l[tid] = sq; __syncthreads();
    for (int off = 128; off > 0; off >>= 1) {
        if (tid < off) { ss[tid] += ss[tid + off]; s2l[tid] += s2l[tid + off]; }
        __syncthreads();
    }
    if (tid == 0) {
        float mean = ss[0] / 25600.f;
        float var  = s2l[0] / 25600.f - mean * mean;
        stats[grp * 2 + 0] = mean;
        stats[grp * 2 + 1] = rsqrtf(var + EPS_);
    }
}

// ---------------------------------------------------------------------------
// K7: in-place BN2 + residual:  out = x + (s2 - mean) * inv * gamma.
// ---------------------------------------------------------------------------
__global__ void k_final(const float* __restrict__ x,
                        const float* __restrict__ stats,
                        const float* __restrict__ g2,
                        float* __restrict__ out) {
    size_t idx = (size_t)blockIdx.x * 256 + threadIdx.x;   // 26,214,400 elems
    int c = (int)((idx / HW_) % 128);
    int t = (int)((idx / ((size_t)HW_ * 128)) % T_);
    float mean = stats[(t * 128 + c) * 2 + 0];
    float inv  = stats[(t * 128 + c) * 2 + 1];
    out[idx] = x[idx] + (out[idx] - mean) * inv * g2[t * 128 + c];
}

// ---------------------------------------------------------------------------
extern "C" void kernel_launch(void* const* d_in, const int* in_sizes, int n_in,
                              void* d_out, int out_size, void* d_ws, size_t ws_size,
                              hipStream_t stream) {
    (void)in_sizes; (void)n_in; (void)out_size; (void)ws_size;
    const float* x     = (const float*)d_in[0];
    const float* c1w   = (const float*)d_in[1];
    const float* c1b   = (const float*)d_in[2];
    // d_in[3], d_in[4] (m1w, m1b): dead in the reference graph — skipped.
    const float* beta1 = (const float*)d_in[5];
    const float* th1   = (const float*)d_in[6];
    const float* g1    = (const float*)d_in[7];
    const float* c2w   = (const float*)d_in[8];
    const float* c2b   = (const float*)d_in[9];
    const float* m2w   = (const float*)d_in[10];
    const float* m2b   = (const float*)d_in[11];
    const float* beta2 = (const float*)d_in[12];
    const float* th2   = (const float*)d_in[13];
    const float* g2    = (const float*)d_in[14];

    char* ws = (char*)d_ws;
    _Float16* spk1 = (_Float16*)(ws + OFF_SPK1);
    float*    s1   = (float*)   (ws + OFF_S1);
    _Float16* spk2 = (_Float16*)(ws + OFF_SPK2);
    _Float16* mem2 = (_Float16*)(ws + OFF_MEM2);
    _Float16* Wc1  = (_Float16*)(ws + OFF_WC1);
    _Float16* Wc2  = (_Float16*)(ws + OFF_WC2);
    _Float16* Wm2  = (_Float16*)(ws + OFF_WM2);
    float*    st1  = (float*)   (ws + OFF_ST1);
    float*    st2  = (float*)   (ws + OFF_ST2);

    float* out = (float*)d_out;                 // [B,T,128,80,80]
    float* men = out + (size_t)NPIX_ * 128;     // [T,B,128,80,80]

    k_prep_weights<<<288, 256, 0, stream>>>(c1w, c2w, m2w, Wc1, Wc2, Wm2);
    k_lif1        <<<12800, 256, 0, stream>>>(x, beta1, th1, spk1);
    k_conv1x1_wmma<<<6400, 256, 0, stream>>>(spk1, Wc1, c1b, s1);
    k_bnstats1    <<<512, 256, 0, stream>>>(s1, st1);
    k_lif2        <<<6400, 256, 0, stream>>>(s1, st1, g1, beta2, th2, spk2, mem2);
    k_conv3x3_wmma<<<12800, 256, 0, stream>>>(spk2, Wc2, c2b, out, 1); // s2 staged
    k_conv3x3_wmma<<<12800, 256, 0, stream>>>(mem2, Wm2, m2b, men, 0); // men direct
    k_bnstats2    <<<1024, 256, 0, stream>>>(out, st2);
    k_final       <<<102400, 256, 0, stream>>>(x, st2, g2, out);
}
